// EGraphSAGE_85152021611246
// MI455X (gfx1250) — compile-verified, matrix operand order checked by reading, not run on hardware
//
#include <hip/hip_runtime.h>

// ---------------------------------------------------------------------------
// EGraphSAGE fused pipeline for gfx1250 (MI455X), wave32 + WMMA bf16 +
// async global->LDS staging (ASYNCcnt).
//   memset cnt+msum -> count scatter -> msg GEMM+scatter (wmma) ->
//   apply GEMM (wmma) -> memset msum -> msg2 (edge feats on the fly) ->
//   apply2 -> predictor GEMM (wmma)
// msum/h1/h2 have N+1 rows: row N is a dump row so scatter/store loops are
// unconditional. A tiles staged in LDS as f32 (async copies bypass VGPRs);
// f32->bf16 conversion happens at fragment-load time, co-executing with WMMA.
// ---------------------------------------------------------------------------

#define D64  64
#define KDIM 128
#define TILE 16
#define WPB  8      // waves per block (256 threads, wave32)

typedef __attribute__((ext_vector_type(16))) __bf16 v16bf;
typedef __attribute__((ext_vector_type(8)))  __bf16 v8bf;
typedef __attribute__((ext_vector_type(8)))  float  v8f;

union ABFrag { v16bf v; v8bf h[2]; };

static __device__ __forceinline__ unsigned short f2bf(float f) {
  __bf16 h = (__bf16)f;
  return __builtin_bit_cast(unsigned short, h);
}

// B fragment from bf16 LDS, per ISA 16-bit 16x32 layout:
//  lanes 0-15 : elems 0..7 = K(k0+0..7),  elems 8..15 = K(k0+16..23)
//  lanes 16-31: elems 0..7 = K(k0+8..15), elems 8..15 = K(k0+24..31)
static __device__ __forceinline__ v16bf load_frag_bf(const unsigned short* row128,
                                                     int k0, int grp) {
  ABFrag f;
  f.h[0] = *(const v8bf*)(row128 + k0 + grp * 8);
  f.h[1] = *(const v8bf*)(row128 + k0 + 16 + grp * 8);
  return f.v;
}

// A fragment from f32 LDS (convert at load; packs to cvt_pk ops)
static __device__ __forceinline__ v16bf load_frag_f32(const float* row128,
                                                      int k0, int grp) {
  const float* p0 = row128 + k0 + grp * 8;
  const float* p1 = row128 + k0 + 16 + grp * 8;
  v16bf r;
  #pragma unroll
  for (int j = 0; j < 8; ++j) {
    r[j]     = (__bf16)p0[j];
    r[j + 8] = (__bf16)p1[j];
  }
  return r;
}

// Async copy of 256B (64 f32): global row-half -> LDS row-half, no VGPR data.
// ISA: the offset applies to both the LDS and global addresses (ASYNCcnt)
static __device__ __forceinline__ void stage_async_256(float* ldsDst,
                                                       const float* gsrc) {
  unsigned l = (unsigned)(unsigned long long)ldsDst;   // LDS byte offset
  #pragma unroll
  for (int j = 0; j < 4; ++j) {
    asm volatile("global_load_async_to_lds_b128 %0, %1, off offset:%2"
                 :: "v"(l), "v"(gsrc), "i"(j * 16) : "memory");
  }
  #pragma unroll
  for (int j = 4; j < 16; ++j) {
    asm volatile("global_load_async_to_lds_b128 %0, %1, off offset:%2"
                 :: "v"(l), "v"(gsrc), "i"(j * 16) : "memory");
  }
}

static __device__ __forceinline__ void wait_async() {
  asm volatile("s_wait_asynccnt 0x0" ::: "memory");
}

// ---------------------------------------------------------------------------
__global__ void count_kernel(const int* __restrict__ dst, float* __restrict__ cnt, int E) {
  int i = blockIdx.x * blockDim.x + threadIdx.x;
  if (i < E) atomicAdd(&cnt[dst[i]], 1.0f);
}

// ---------------------------------------------------------------------------
// Message GEMM + scatter-add into msum[dst] (msum has N+1 rows; row N = dump).
// mode 0: A = [hfeat[src[e]] | efeats[e]]              (both halves async)
// mode 1: A = [hfeat[src[e]] | 0.5*(hfeat[src]+hfeat[dst])] (2nd half manual)
__global__ void __launch_bounds__(256)
msg_kernel(const float* __restrict__ hfeat,   // [N,64]
           const float* __restrict__ efeats,  // [E,64] (mode 0) or unused
           const int*   __restrict__ src,
           const int*   __restrict__ dst,
           const float* __restrict__ W,       // [128,64] row-major (k,n)
           const float* __restrict__ b,       // [64]
           float*       __restrict__ msum,    // [N+1,64]
           int N, int E, int mode)
{
  __shared__ __align__(16) unsigned short wt[D64][KDIM];   // W^T bf16
  __shared__ float bs[D64];
  __shared__ __align__(16) float aTf[WPB][TILE][KDIM];     // f32 A tiles
  __shared__ int dIdx[WPB][TILE];

  const int tid = threadIdx.x;
  for (int i = tid; i < D64 * KDIM; i += blockDim.x) {
    int n = i >> 7, k = i & 127;
    wt[n][k] = f2bf(W[k * D64 + n]);
  }
  if (tid < D64) bs[tid] = b[tid];
  __syncthreads();

  const int wave = tid >> 5;
  const int lane = tid & 31;
  const int row  = lane & 15;
  const int grp  = lane >> 4;

  float bias4[4];
  v16bf Bf[4][4];                               // tile-invariant B fragments
  #pragma unroll
  for (int nt = 0; nt < 4; ++nt) {
    bias4[nt] = bs[nt * 16 + row];
    #pragma unroll
    for (int kk = 0; kk < 4; ++kk)
      Bf[nt][kk] = load_frag_bf(&wt[nt * 16 + row][0], kk * 32, grp);
  }

  const int nTiles = (E + TILE - 1) / TILE;
  const int nSuper = (nTiles + WPB - 1) / WPB;

  for (int bt = blockIdx.x; bt < nSuper; bt += gridDim.x) {
    const int t       = bt * WPB + wave;
    const bool active = (t < nTiles);          // wave-uniform
    const int base    = t * TILE;

    if (active) {
      __builtin_prefetch(src + base + TILE * WPB, 0, 1);
      __builtin_prefetch(dst + base + TILE * WPB, 0, 1);
      int r  = lane >> 1;
      int hf = lane & 1;
      int e0 = base + r;
      int e  = (e0 < E) ? e0 : (E - 1);
      int s  = src[e];
      if (hf == 0) {
        stage_async_256(&aTf[wave][r][0], hfeat + (size_t)s * D64);
      } else {
        int dn = dst[e];
        dIdx[wave][r] = (e0 < E) ? dn : N;     // OOB edges -> dump row
        if (mode == 0) {
          stage_async_256(&aTf[wave][r][64], efeats + (size_t)e * D64);
        } else {
          const float4* ps = (const float4*)(hfeat + (size_t)s  * D64);
          const float4* pd = (const float4*)(hfeat + (size_t)dn * D64);
          #pragma unroll
          for (int j = 0; j < 16; ++j) {
            float4 a = ps[j], c = pd[j];
            float4 o; o.x = 0.5f*(a.x+c.x); o.y = 0.5f*(a.y+c.y);
                      o.z = 0.5f*(a.z+c.z); o.w = 0.5f*(a.w+c.w);
            *(float4*)&aTf[wave][r][64 + 4*j] = o;
          }
        }
      }
    }
    __syncthreads();

    if (active) {
      wait_async();                             // our async slice is landed
      const float* arow = &aTf[wave][row][0];

      v16bf Af[4];
      #pragma unroll
      for (int kk = 0; kk < 4; ++kk) Af[kk] = load_frag_f32(arow, kk * 32, grp);

      unsigned off8[8];
      #pragma unroll
      for (int i = 0; i < 8; ++i)
        off8[i] = (unsigned)dIdx[wave][i + grp * 8] * (unsigned)D64 + (unsigned)row;

      #pragma unroll
      for (int nt = 0; nt < 4; ++nt) {
        v8f c;
        #pragma unroll
        for (int i = 0; i < 8; ++i) c[i] = bias4[nt];
        #pragma unroll
        for (int kk = 0; kk < 4; ++kk)
          c = __builtin_amdgcn_wmma_f32_16x16x32_bf16(false, Af[kk], false, Bf[nt][kk],
                                                      (short)0, c, false, false);
        #pragma unroll
        for (int i = 0; i < 8; ++i)
          atomicAdd(msum + off8[i] + nt * 16, c[i]);   // unconditional
      }
    }
    __syncthreads();
  }
}

// ---------------------------------------------------------------------------
// Apply GEMM: hout = relu([hin | msum/max(cnt,1)] @ W + b); hout has N+1 rows.
__global__ void __launch_bounds__(256)
apply_kernel(const float* __restrict__ hin,   // [N,64]
             const float* __restrict__ msum,  // [N+1,64]
             const float* __restrict__ cnt,   // [N]
             const float* __restrict__ W,     // [128,64]
             const float* __restrict__ b,     // [64]
             float*       __restrict__ hout,  // [N+1,64]
             int N)
{
  __shared__ __align__(16) unsigned short wt[D64][KDIM];
  __shared__ float bs[D64];
  __shared__ __align__(16) float aTf[WPB][TILE][KDIM];

  const int tid = threadIdx.x;
  for (int i = tid; i < D64 * KDIM; i += blockDim.x) {
    int n = i >> 7, k = i & 127;
    wt[n][k] = f2bf(W[k * D64 + n]);
  }
  if (tid < D64) bs[tid] = b[tid];
  __syncthreads();

  const int wave = tid >> 5;
  const int lane = tid & 31;
  const int row  = lane & 15;
  const int grp  = lane >> 4;

  float bias4[4];
  v16bf Bf[4][4];
  #pragma unroll
  for (int nt = 0; nt < 4; ++nt) {
    bias4[nt] = bs[nt * 16 + row];
    #pragma unroll
    for (int kk = 0; kk < 4; ++kk)
      Bf[nt][kk] = load_frag_bf(&wt[nt * 16 + row][0], kk * 32, grp);
  }

  const int nTiles = (N + TILE - 1) / TILE;
  const int nSuper = (nTiles + WPB - 1) / WPB;

  for (int bt = blockIdx.x; bt < nSuper; bt += gridDim.x) {
    const int t       = bt * WPB + wave;
    const bool active = (t < nTiles);
    const int base    = t * TILE;

    if (active) {
      int r  = lane >> 1;
      int hf = lane & 1;
      int n0 = base + r;
      int n  = (n0 < N) ? n0 : (N - 1);
      if (hf == 0) {
        stage_async_256(&aTf[wave][r][0], hin + (size_t)n * D64);
      } else {
        float cv  = cnt[n];
        float inv = 1.0f / fmaxf(cv, 1.0f);
        const float4* p = (const float4*)(msum + (size_t)n * D64);
        #pragma unroll
        for (int j = 0; j < 16; ++j) {
          float4 v = p[j];
          float4 o; o.x = v.x*inv; o.y = v.y*inv; o.z = v.z*inv; o.w = v.w*inv;
          *(float4*)&aTf[wave][r][64 + 4*j] = o;
        }
      }
    }
    __syncthreads();

    if (active) {
      wait_async();
      const float* arow = &aTf[wave][row][0];
      v16bf Af[4];
      #pragma unroll
      for (int kk = 0; kk < 4; ++kk) Af[kk] = load_frag_f32(arow, kk * 32, grp);

      unsigned off8[8];
      #pragma unroll
      for (int i = 0; i < 8; ++i) {
        int n0 = base + i + grp * 8;
        unsigned nc = (n0 < N) ? (unsigned)n0 : (unsigned)N;   // OOB -> dump row
        off8[i] = nc * (unsigned)D64 + (unsigned)row;
      }

      #pragma unroll
      for (int nt = 0; nt < 4; ++nt) {
        v8f c;
        #pragma unroll
        for (int i = 0; i < 8; ++i) c[i] = bias4[nt];
        #pragma unroll
        for (int kk = 0; kk < 4; ++kk)
          c = __builtin_amdgcn_wmma_f32_16x16x32_bf16(false, Af[kk], false, Bf[nt][kk],
                                                      (short)0, c, false, false);
        #pragma unroll
        for (int i = 0; i < 8; ++i)
          hout[off8[i] + nt * 16] = fmaxf(c[i], 0.0f);         // unconditional
      }
    }
    __syncthreads();
  }
}

// ---------------------------------------------------------------------------
// Edge predictor: out[e, 0..10] = [h2[src]|h2[dst]] @ Wpred + bpred
__global__ void __launch_bounds__(256)
pred_kernel(const float* __restrict__ h2,    // [N+1,64]
            const int*   __restrict__ src,
            const int*   __restrict__ dst,
            const float* __restrict__ Wp,    // [128,11]
            const float* __restrict__ bp,    // [11]
            float*       __restrict__ out,   // [E,11]
            float*       __restrict__ dump,  // scratch for OOB lanes
            int E)
{
  __shared__ __align__(16) unsigned short wt[16][KDIM];   // padded to 16 cols
  __shared__ float bs[16];
  __shared__ __align__(16) float aTf[WPB][TILE][KDIM];

  const int tid = threadIdx.x;
  for (int i = tid; i < 16 * KDIM; i += blockDim.x) {
    int n = i >> 7, k = i & 127;
    wt[n][k] = (n < 11) ? f2bf(Wp[k * 11 + n]) : (unsigned short)0;
  }
  if (tid < 16) bs[tid] = (tid < 11) ? bp[tid] : 0.0f;
  __syncthreads();

  const int wave = tid >> 5;
  const int lane = tid & 31;
  const int row  = lane & 15;
  const int grp  = lane >> 4;

  const float bias = bs[row];
  v16bf Bf[4];
  #pragma unroll
  for (int kk = 0; kk < 4; ++kk)
    Bf[kk] = load_frag_bf(&wt[row][0], kk * 32, grp);

  const int nTiles = (E + TILE - 1) / TILE;
  const int nSuper = (nTiles + WPB - 1) / WPB;

  for (int bt = blockIdx.x; bt < nSuper; bt += gridDim.x) {
    const int t       = bt * WPB + wave;
    const bool active = (t < nTiles);
    const int base    = t * TILE;
    const bool full   = active && (base + TILE <= E);

    if (active) {
      int r  = lane >> 1;
      int hf = lane & 1;
      int e0 = base + r;
      int e  = (e0 < E) ? e0 : (E - 1);
      int n  = (hf == 0) ? src[e] : dst[e];
      stage_async_256(&aTf[wave][r][hf * 64], h2 + (size_t)n * D64);
    }
    __syncthreads();

    if (active) {
      wait_async();
      const int col = row;        // single 16-wide N tile (11 valid)
      v8f c;
      #pragma unroll
      for (int i = 0; i < 8; ++i) c[i] = bias;
      const float* arow = &aTf[wave][row][0];
      v16bf Af[4];
      #pragma unroll
      for (int kk = 0; kk < 4; ++kk) Af[kk] = load_frag_f32(arow, kk * 32, grp);
      #pragma unroll
      for (int kk = 0; kk < 4; ++kk)
        c = __builtin_amdgcn_wmma_f32_16x16x32_bf16(false, Af[kk], false, Bf[kk],
                                                    (short)0, c, false, false);
      if (col < 11) {
        if (full) {
          #pragma unroll
          for (int i = 0; i < 8; ++i) {
            int e = base + i + grp * 8;
            out[(size_t)e * 11 + col] = c[i];
          }
        } else {
          #pragma unroll
          for (int i = 0; i < 8; ++i) {
            int e = base + i + grp * 8;
            float* p = (e < E) ? (out + (size_t)e * 11 + col) : dump;  // branchless
            *p = c[i];
          }
        }
      }
    }
    __syncthreads();
  }
}

// ---------------------------------------------------------------------------
extern "C" void kernel_launch(void* const* d_in, const int* in_sizes, int n_in,
                              void* d_out, int out_size, void* d_ws, size_t ws_size,
                              hipStream_t stream) {
  const float* nfeats = (const float*)d_in[0];
  const float* efeats = (const float*)d_in[1];
  const int*   src    = (const int*)d_in[2];
  const int*   dst    = (const int*)d_in[3];
  const float* Wm1 = (const float*)d_in[4];  const float* bm1 = (const float*)d_in[5];
  const float* Wa1 = (const float*)d_in[6];  const float* ba1 = (const float*)d_in[7];
  const float* Wm2 = (const float*)d_in[8];  const float* bm2 = (const float*)d_in[9];
  const float* Wa2 = (const float*)d_in[10]; const float* ba2 = (const float*)d_in[11];
  const float* Wp  = (const float*)d_in[12]; const float* bpr = (const float*)d_in[13];
  float* out = (float*)d_out;

  const int N = in_sizes[0] / D64;
  const int E = in_sizes[2];
  const size_t ROWS = (size_t)(N + 1) * D64;   // +1 dump row

  float* cnt  = (float*)d_ws;          // [N]
  float* msum = cnt + N;               // [N+1,64]
  float* h1   = msum + ROWS;           // [N+1,64]
  float* h2   = h1   + ROWS;           // [N+1,64]
  float* dump = h2   + ROWS;           // [64]

  const int nTilesE = (E + TILE - 1) / TILE;
  const int nTilesN = (N + TILE - 1) / TILE;
  int gE = (nTilesE + WPB - 1) / WPB; if (gE > 8192) gE = 8192;
  int gN = (nTilesN + WPB - 1) / WPB; if (gN > 8192) gN = 8192;

  // zero cnt + msum (contiguous)
  hipMemsetAsync(cnt, 0, ((size_t)N + ROWS) * sizeof(float), stream);
  count_kernel<<<(E + 255) / 256, 256, 0, stream>>>(dst, cnt, E);

  // layer 1
  msg_kernel<<<gE, 256, 0, stream>>>(nfeats, efeats, src, dst, Wm1, bm1, msum, N, E, 0);
  apply_kernel<<<gN, 256, 0, stream>>>(nfeats, msum, cnt, Wa1, ba1, h1, N);

  // layer 2 (edge feats recomputed on the fly: 0.5*(h1[src]+h1[dst]))
  hipMemsetAsync(msum, 0, ROWS * sizeof(float), stream);
  msg_kernel<<<gE, 256, 0, stream>>>(h1, nullptr, src, dst, Wm2, bm2, msum, N, E, 1);
  apply_kernel<<<gN, 256, 0, stream>>>(h1, msum, cnt, Wa2, ba2, h2, N);

  // predictor
  pred_kernel<<<gE, 256, 0, stream>>>(h2, src, dst, Wp, bpr, out, dump, E);
}
